// MultiHeadAttention_1795296329949
// MI455X (gfx1250) — matrix operand order repaired and checked by reading, not compiled, not run on hardware
//
#include <hip/hip_runtime.h>
#include <cstdint>
#include <cstddef>

// ---------------------------------------------------------------------------
// Problem constants (reference: B=2, S=2048, D=1024, H=16, HD=64, ROT=32)
// ---------------------------------------------------------------------------
#define Bn   2
#define Sn   2048
#define Dn   1024
#define Hn   16
#define HDn  64
#define ROTn 32
#define Mrows (Bn * Sn)            // 4096 rows for the projection GEMMs

typedef __bf16 bf16_t;
typedef __attribute__((ext_vector_type(16))) __bf16   v16bf;
typedef __attribute__((ext_vector_type(8)))  float    v8f;
typedef __attribute__((ext_vector_type(4)))  uint32_t v4u;

union Frag32B { v16bf v; v4u q[2]; };

__device__ __forceinline__ v16bf load2(const unsigned short* p0, const unsigned short* p1) {
  Frag32B f;
  f.q[0] = *(const v4u*)p0;   // 16 bytes
  f.q[1] = *(const v4u*)p1;   // 16 bytes
  return f.v;
}

__device__ __forceinline__ unsigned short f2bf(float x) {
  uint32_t u = __float_as_uint(x);
  u += 0x7fffu + ((u >> 16) & 1u);          // round-to-nearest-even
  return (unsigned short)(u >> 16);
}
__device__ __forceinline__ float bf2f(unsigned short s) {
  return __uint_as_float(((uint32_t)s) << 16);
}

#define WMMA_BF16(a, b, c) \
  __builtin_amdgcn_wmma_f32_16x16x32_bf16(false, (a), false, (b), (short)0, (c), false, false)

// ---------------------------------------------------------------------------
// fp32 -> bf16 convert
// ---------------------------------------------------------------------------
__global__ void cvt_f32_bf16(const float* __restrict__ in, unsigned short* __restrict__ out, int n) {
  int i = blockIdx.x * blockDim.x + threadIdx.x;
  if (i < n) out[i] = f2bf(in[i]);
}

// ---------------------------------------------------------------------------
// GEMM: Y[M,N] = X[M,K] @ W[N,K]^T  (bf16 in, fp32 accum, bf16 or fp32 out)
// Block = 128 threads (4 waves). Each wave owns a 16(M) x 64(N) tile.
// Workgroup covers 64x64. K stepped by 32 with v_wmma_f32_16x16x32_bf16.
// ---------------------------------------------------------------------------
template <bool OUT_F32>
__global__ void __launch_bounds__(128)
gemm_xwt(const unsigned short* __restrict__ X, const unsigned short* __restrict__ W,
         void* __restrict__ Y, int M, int N, int K)
{
  const int ntn  = N >> 6;
  const int tile = blockIdx.x;
  const int tm   = tile / ntn;
  const int tn   = tile - tm * ntn;
  const int wave = threadIdx.x >> 5;
  const int lane = threadIdx.x & 31;
  const int l16  = lane & 15;
  const int hi   = lane >> 4;         // 0 for lanes 0-15, 1 for 16-31

  const int m0 = tm * 64 + wave * 16;
  const int n0 = tn * 64;

  const unsigned short* xrow = X + (size_t)(m0 + l16) * K;
  const unsigned short* wr0  = W + (size_t)(n0 +  0 + l16) * K;
  const unsigned short* wr1  = W + (size_t)(n0 + 16 + l16) * K;
  const unsigned short* wr2  = W + (size_t)(n0 + 32 + l16) * K;
  const unsigned short* wr3  = W + (size_t)(n0 + 48 + l16) * K;

  v8f acc0 = {}, acc1 = {}, acc2 = {}, acc3 = {};

  for (int k0 = 0; k0 < K; k0 += 32) {
    // A frag: row m0+l16, 16 contiguous K values per lane (lo lanes k0..k0+15, hi lanes k0+16..+31)
    v16bf a = load2(xrow + k0 + hi * 16, xrow + k0 + 8 + hi * 16);
    // B frags: 16 contiguous K values per lane (lo lanes k0..k0+15, hi lanes k0+16..+31)
    const int kb = k0 + hi * 16;
    v16bf b0 = load2(wr0 + kb, wr0 + kb + 8);
    v16bf b1 = load2(wr1 + kb, wr1 + kb + 8);
    v16bf b2 = load2(wr2 + kb, wr2 + kb + 8);
    v16bf b3 = load2(wr3 + kb, wr3 + kb + 8);
    acc0 = WMMA_BF16(a, b0, acc0);
    acc1 = WMMA_BF16(a, b1, acc1);
    acc2 = WMMA_BF16(a, b2, acc2);
    acc3 = WMMA_BF16(a, b3, acc3);
  }

  // C/D layout: vgpr j, lanes 0-15 -> M=j ; lanes 16-31 -> M=j+8 ; N = lane%16
  v8f accs[4] = {acc0, acc1, acc2, acc3};
#pragma unroll
  for (int t = 0; t < 4; ++t) {
#pragma unroll
    for (int j = 0; j < 8; ++j) {
      const int row = m0 + j + hi * 8;
      const int col = n0 + t * 16 + l16;
      const float v = accs[t][j];
      if (OUT_F32) ((float*)Y)[(size_t)row * N + col] = v;
      else         ((unsigned short*)Y)[(size_t)row * N + col] = f2bf(v);
    }
  }
}

// ---------------------------------------------------------------------------
// RoPE (interleaved pairs) applied in place to bf16 Q and K.
// One thread per (b,s,h,r) pair; total = B*S*H*ROT = 2^21.
// ---------------------------------------------------------------------------
__global__ void rope_qk(unsigned short* __restrict__ Q, unsigned short* __restrict__ Kk,
                        const float* __restrict__ freqs)
{
  const int i = blockIdx.x * blockDim.x + threadIdx.x;
  if (i >= Bn * Sn * Hn * ROTn) return;
  const int r = i & (ROTn - 1);
  const int h = (i >> 5) & (Hn - 1);
  const int s = (i >> 9) & (Sn - 1);
  const int b = i >> 20;

  const float c  = freqs[(s * ROTn + r) * 2 + 0];
  const float sn = freqs[(s * ROTn + r) * 2 + 1];
  const size_t base = (size_t)(b * Sn + s) * Dn + h * HDn + r * 2;

  {
    const float x0 = bf2f(Q[base]), x1 = bf2f(Q[base + 1]);
    Q[base]     = f2bf(x0 * c - x1 * sn);
    Q[base + 1] = f2bf(x1 * c + x0 * sn);
  }
  {
    const float x0 = bf2f(Kk[base]), x1 = bf2f(Kk[base + 1]);
    Kk[base]     = f2bf(x0 * c - x1 * sn);
    Kk[base + 1] = f2bf(x1 * c + x0 * sn);
  }
}

// ---------------------------------------------------------------------------
// V transpose: V[b*S+s][h*HD+d]  ->  Vt[((b*H+h)*HD+d)*S + s]
// ---------------------------------------------------------------------------
__global__ void v_transpose(const unsigned short* __restrict__ V, unsigned short* __restrict__ Vt)
{
  const int i = blockIdx.x * blockDim.x + threadIdx.x;
  if (i >= Bn * Hn * HDn * Sn) return;
  const int s = i & (Sn - 1);
  const int d = (i >> 11) & (HDn - 1);
  const int h = (i >> 17) & (Hn - 1);
  const int b = i >> 21;
  Vt[i] = V[(size_t)(b * Sn + s) * Dn + h * HDn + d];
}

// ---------------------------------------------------------------------------
// Flash attention, non-causal.  Grid = B*H*(S/64) workgroups of 128 threads.
// Each wave: 16 query rows; workgroup: 64 query rows of one (b,h).
// Iterate 32 key blocks of 64 keys with online softmax.
// ---------------------------------------------------------------------------
__global__ void __launch_bounds__(128)
flash_attn(const unsigned short* __restrict__ Q, const unsigned short* __restrict__ K,
           const unsigned short* __restrict__ Vt, unsigned short* __restrict__ O)
{
  __shared__ unsigned short ldsP[4][16 * 64];   // per-wave P staging (C-layout -> A-layout)

  const int wave = threadIdx.x >> 5;
  const int lane = threadIdx.x & 31;
  const int l16  = lane & 15;
  const int hi   = lane >> 4;

  const int blk  = blockIdx.x;
  const int qblk = blk & 31;          // S/64 = 32
  const int h    = (blk >> 5) & (Hn - 1);
  const int b    = blk >> 9;

  const int m0 = qblk * 64 + wave * 16;           // first query row of this wave

  // Q A-frags for both d-chunks (d0 = 0 and 32), loaded once
  const unsigned short* qrow = Q + (size_t)(b * Sn + m0 + l16) * Dn + h * HDn;
  const v16bf aq0 = load2(qrow +  0 + hi * 16, qrow +  8 + hi * 16);
  const v16bf aq1 = load2(qrow + 32 + hi * 16, qrow + 40 + hi * 16);

  const unsigned short* kbase  = K  + (size_t)(b * Sn) * Dn + h * HDn;
  const unsigned short* vtbase = Vt + (size_t)((b * Hn + h) * HDn) * Sn;
  unsigned short* myP = ldsP[wave];

  float mrow[8], lrow[8];
  v8f acc[4];
#pragma unroll
  for (int j = 0; j < 8; ++j) { mrow[j] = -3.0e38f; lrow[j] = 0.0f; }
#pragma unroll
  for (int t = 0; t < 4; ++t) acc[t] = (v8f){};

  const float scale = 0.125f;   // HD^-0.5

  for (int kb = 0; kb < Sn / 64; ++kb) {
    const int n0 = kb * 64;

    // ---- scores = Q @ K^T (16 x 64 per wave) ----
    v8f s_acc[4];
#pragma unroll
    for (int t = 0; t < 4; ++t) {
      s_acc[t] = (v8f){};
      const unsigned short* krow = kbase + (size_t)(n0 + t * 16 + l16) * Dn;
      const v16bf bk0 = load2(krow +  0 + hi * 16, krow +  8 + hi * 16);
      const v16bf bk1 = load2(krow + 32 + hi * 16, krow + 40 + hi * 16);
      s_acc[t] = WMMA_BF16(aq0, bk0, s_acc[t]);
      s_acc[t] = WMMA_BF16(aq1, bk1, s_acc[t]);
    }

    // ---- online softmax over this 16x64 block ----
    float mnew[8], alpha[8], lsum[8];
#pragma unroll
    for (int j = 0; j < 8; ++j) {
      float v = s_acc[0][j] * scale;
#pragma unroll
      for (int t = 1; t < 4; ++t) v = fmaxf(v, s_acc[t][j] * scale);
      // row max across the 16 lanes of this half-wave
#pragma unroll
      for (int off = 1; off < 16; off <<= 1) v = fmaxf(v, __shfl_xor(v, off, 32));
      mnew[j]  = fmaxf(mrow[j], v);
      alpha[j] = __expf(mrow[j] - mnew[j]);
      mrow[j]  = mnew[j];
      lsum[j]  = 0.0f;
    }
#pragma unroll
    for (int t = 0; t < 4; ++t) {
#pragma unroll
      for (int j = 0; j < 8; ++j) {
        const float p = __expf(s_acc[t][j] * scale - mnew[j]);
        s_acc[t][j] = p;
        lsum[j] += p;
      }
    }
#pragma unroll
    for (int j = 0; j < 8; ++j) {
      float v = lsum[j];
#pragma unroll
      for (int off = 1; off < 16; off <<= 1) v += __shfl_xor(v, off, 32);
      lrow[j] = lrow[j] * alpha[j] + v;
    }
#pragma unroll
    for (int t = 0; t < 4; ++t)
#pragma unroll
      for (int j = 0; j < 8; ++j) acc[t][j] *= alpha[j];

    // ---- P: C-layout -> bf16 A-layout via per-wave LDS tile ----
#pragma unroll
    for (int t = 0; t < 4; ++t) {
#pragma unroll
      for (int j = 0; j < 8; ++j) {
        const int m = j + hi * 8;
        myP[m * 64 + t * 16 + l16] = f2bf(s_acc[t][j]);
      }
    }
    asm volatile("s_wait_dscnt 0x0" ::: "memory");   // wave-local LDS RAW fence

    const unsigned short* prow = myP + l16 * 64;
    const v16bf ap0 = load2(prow +  0 + hi * 16, prow +  8 + hi * 16);
    const v16bf ap1 = load2(prow + 32 + hi * 16, prow + 40 + hi * 16);

    // ---- O += P @ V  (B from Vt: contiguous along key index) ----
#pragma unroll
    for (int t = 0; t < 4; ++t) {
      const unsigned short* vrow = vtbase + (size_t)(t * 16 + l16) * Sn + n0;
      const v16bf bv0 = load2(vrow +  0 + hi * 16, vrow +  8 + hi * 16);
      const v16bf bv1 = load2(vrow + 32 + hi * 16, vrow + 40 + hi * 16);
      acc[t] = WMMA_BF16(ap0, bv0, acc[t]);
      acc[t] = WMMA_BF16(ap1, bv1, acc[t]);
    }
  }

  // ---- epilogue: normalize and store bf16 O[b][s][h*HD + d] ----
  float inv[8];
#pragma unroll
  for (int j = 0; j < 8; ++j) inv[j] = 1.0f / lrow[j];
#pragma unroll
  for (int t = 0; t < 4; ++t) {
#pragma unroll
    for (int j = 0; j < 8; ++j) {
      const int row = m0 + j + hi * 8;
      const int col = h * HDn + t * 16 + l16;
      O[(size_t)(b * Sn + row) * Dn + col] = f2bf(acc[t][j] * inv[j]);
    }
  }
}

// ---------------------------------------------------------------------------
// Host-side launch
// ---------------------------------------------------------------------------
extern "C" void kernel_launch(void* const* d_in, const int* in_sizes, int n_in,
                              void* d_out, int out_size, void* d_ws, size_t ws_size,
                              hipStream_t stream)
{
  (void)in_sizes; (void)n_in; (void)out_size; (void)ws_size;

  const float* hidden = (const float*)d_in[0];   // (B,S,D)
  const float* freqs  = (const float*)d_in[1];   // (S,ROT,2)
  const float* wq     = (const float*)d_in[2];   // (D,D)
  const float* wk     = (const float*)d_in[3];
  const float* wv     = (const float*)d_in[4];
  const float* wo     = (const float*)d_in[5];
  float*       out    = (float*)d_out;           // (B,S,D) fp32

  const size_t nH = (size_t)Bn * Sn * Dn;        // 4,194,304
  const size_t nW = (size_t)Dn * Dn;             // 1,048,576

  unsigned short* Hb  = (unsigned short*)d_ws;
  unsigned short* Wqb = Hb  + nH;
  unsigned short* Wkb = Wqb + nW;
  unsigned short* Wvb = Wkb + nW;
  unsigned short* Wob = Wvb + nW;
  unsigned short* Qb  = Wob + nW;
  unsigned short* Kb  = Qb  + nH;
  unsigned short* Vb  = Kb  + nH;
  unsigned short* Vtb = Vb  + nH;
  unsigned short* Ob  = Vtb + nH;
  // total: 6*nH + 4*nW bf16 = ~56 MB

  // 1) fp32 -> bf16 converts
  cvt_f32_bf16<<<(int)((nH + 255) / 256), 256, 0, stream>>>(hidden, Hb, (int)nH);
  cvt_f32_bf16<<<(int)((nW + 255) / 256), 256, 0, stream>>>(wq, Wqb, (int)nW);
  cvt_f32_bf16<<<(int)((nW + 255) / 256), 256, 0, stream>>>(wk, Wkb, (int)nW);
  cvt_f32_bf16<<<(int)((nW + 255) / 256), 256, 0, stream>>>(wv, Wvb, (int)nW);
  cvt_f32_bf16<<<(int)((nW + 255) / 256), 256, 0, stream>>>(wo, Wob, (int)nW);

  // 2) Q/K/V projections: (4096,1024) x (1024,1024)^T, 64x64 tiles -> 1024 blocks
  const int gemm_blocks = (Mrows / 64) * (Dn / 64);
  gemm_xwt<false><<<gemm_blocks, 128, 0, stream>>>(Hb, Wqb, Qb, Mrows, Dn, Dn);
  gemm_xwt<false><<<gemm_blocks, 128, 0, stream>>>(Hb, Wkb, Kb, Mrows, Dn, Dn);
  gemm_xwt<false><<<gemm_blocks, 128, 0, stream>>>(Hb, Wvb, Vb, Mrows, Dn, Dn);

  // 3) RoPE on Q and K
  const int rope_total = Bn * Sn * Hn * ROTn;    // 2^21
  rope_qk<<<rope_total / 256, 256, 0, stream>>>(Qb, Kb, freqs);

  // 4) transpose V -> Vt[b][h][d][s]
  const int vt_total = Bn * Hn * HDn * Sn;       // 2^22
  v_transpose<<<vt_total / 256, 256, 0, stream>>>(Vb, Vtb);

  // 5) flash attention: B*H*(S/64) = 1024 workgroups
  flash_attn<<<Bn * Hn * (Sn / 64), 128, 0, stream>>>(Qb, Kb, Vtb, Ob);

  // 6) output projection -> fp32 d_out
  gemm_xwt<true><<<gemm_blocks, 128, 0, stream>>>(Ob, Wob, (void*)out, Mrows, Dn, Dn);
}
